// ImplicitNetworkGrid_87754771792671
// MI455X (gfx1250) — matrix operand order repaired and checked by reading, not compiled, not run on hardware
//
#include <hip/hip_runtime.h>
#include <hip/hip_bf16.h>
#include <math.h>

// ---------------------------------------------------------------------------
// ImplicitNetworkGrid for MI455X (gfx1250, wave32, WMMA)
//   Kernel 1: convert/pad MLP weights f32 -> bf16 (row-major [Npad][256])
//   Kernel 2: features (hash-grid encode + streaming ball query + posenc)
//   Kernel 3: fused 9-layer MLP, activations resident in LDS,
//             v_wmma_f32_16x16x32_bf16 for every GEMM, branch-free softplus.
// ---------------------------------------------------------------------------

typedef __bf16 bf16_t;
typedef __attribute__((ext_vector_type(16))) __bf16 v16bf;
typedef __attribute__((ext_vector_type(8)))  __bf16 v8bf;
typedef __attribute__((ext_vector_type(8)))  float  v8f;

#define NPOINTS   16384        // 512 * 32
#define MPART     4096
#define KNN       32
#define NLEVELS   16
#define LOG2HASH  19
#define HASHMAP   (1u << LOG2HASH)
#define RADIUS    0.1f
#define INVSQRT2  0.70710678118654752440f

// ---------------------------------------------------------------------------
// Weight conversion: f32 [N][K] -> bf16 [Npad][256], zero padded
// ---------------------------------------------------------------------------
__global__ __launch_bounds__(256) void convert_w_kernel(
    const float* __restrict__ W, bf16_t* __restrict__ dst,
    int N, int K, int Npad)
{
    int idx = blockIdx.x * 256 + threadIdx.x;
    if (idx >= Npad * 256) return;
    int n = idx >> 8;
    int k = idx & 255;
    float v = (n < N && k < K) ? W[n * K + k] : 0.0f;
    dst[idx] = (bf16_t)v;
}

// ---------------------------------------------------------------------------
// posenc writer: [x, sin(2^0 x), cos(2^0 x), ..., sin(2^{F-1}x), cos(2^{F-1}x)]
// ---------------------------------------------------------------------------
__device__ __forceinline__ void write_posenc(bf16_t* __restrict__ row, int off,
                                             const float* v, int C, int F)
{
    for (int c = 0; c < C; ++c) row[off + c] = (bf16_t)v[c];
    int o = off + C;
    float fr = 1.0f;
    for (int f = 0; f < F; ++f) {
        for (int c = 0; c < C; ++c) row[o++] = (bf16_t)__sinf(v[c] * fr);
        for (int c = 0; c < C; ++c) row[o++] = (bf16_t)__cosf(v[c] * fr);
        fr *= 2.0f;
    }
}

// ---------------------------------------------------------------------------
// Feature kernel: one thread per point. Particles staged in LDS (48 KB,
// broadcast reads). Ball query is streamed (first 32 in index order).
// Writes x_in as bf16 [NPOINTS][256] (cols 230..255 zero).
// ---------------------------------------------------------------------------
__global__ __launch_bounds__(256) void feature_kernel(
    const float* __restrict__ inp,       // [16384,3]
    const float* __restrict__ particles, // [4096,3]
    const float* __restrict__ table,     // [16][2^19][2]
    bf16_t* __restrict__ xin)            // [16384,256]
{
    __shared__ float sp[MPART * 3];
    for (int i = threadIdx.x; i < MPART * 3; i += 256) sp[i] = particles[i];
    __syncthreads();

    const int p = blockIdx.x * 256 + threadIdx.x;
    const float x0 = inp[p * 3 + 0];
    const float x1 = inp[p * 3 + 1];
    const float x2 = inp[p * 3 + 2];

    // ---- streaming ball query -------------------------------------------
    const float R2 = RADIUS * RADIUS;
    const float invR = 1.0f / RADIUS;
    int   cnt = 0;
    float wsum = 0.f;
    float sw0 = 0.f, sw1 = 0.f, sw2 = 0.f;   // sum w * nn
    float sd0 = 0.f, sd1 = 0.f, sd2 = 0.f;   // sum diff
    float sq0 = 0.f, sq1 = 0.f, sq2 = 0.f;   // sum diff^2
    for (int m = 0; m < MPART; ++m) {
        float px = sp[3 * m + 0], py = sp[3 * m + 1], pz = sp[3 * m + 2];
        float dx = px - x0, dy = py - x1, dz = pz - x2;
        float d2 = dx * dx + dy * dy + dz * dz;
        if (cnt < KNN && d2 < R2) {
            ++cnt;
            float d = sqrtf(fmaxf(d2, 1e-24f));
            float t = d * invR;
            float w = fmaxf(1.0f - t * t * t, 0.0f);
            wsum += w;
            sw0 += w * px; sw1 += w * py; sw2 += w * pz;
            sd0 += dx; sd1 += dy; sd2 += dz;
            sq0 += dx * dx; sq1 += dy * dy; sq2 += dz * dz;
        }
    }
    // invalid slots: nn = 0 -> diff = -inp, contributes w(|inp|) to wsum only
    {
        float dinv = sqrtf(fmaxf(x0 * x0 + x1 * x1 + x2 * x2, 1e-24f));
        float ti = dinv * invR;
        float winv = fmaxf(1.0f - ti * ti * ti, 0.0f);
        wsum += (float)(KNN - cnt) * winv;
    }
    float inv_ws = 1.0f / (wsum + 1e-12f);
    float sm[3] = { sw0 * inv_ws, sw1 * inv_ws, sw2 * inv_ws };
    float invn = 1.0f / ((float)cnt + 1e-12f);
    float vm0 = sd0 * invn, vm1 = sd1 * invn, vm2 = sd2 * invn;
    float var[3] = {
        (sq0 - 2.f * vm0 * sd0 + (float)cnt * vm0 * vm0) * invn,
        (sq1 - 2.f * vm1 * sd1 + (float)cnt * vm1 * vm1) * invn,
        (sq2 - 2.f * vm2 * sd2 + (float)cnt * vm2 * vm2) * invn };

    bf16_t* row = xin + (size_t)p * 256;

    // ---- embeddings ------------------------------------------------------
    float xv[3] = { x0, x1, x2 };
    write_posenc(row, 0, xv, 3, 10);          // hit_emb    [0, 63)
    float wv[1] = { wsum };
    write_posenc(row, 63, wv, 1, 4);          // density    [63, 72)
    write_posenc(row, 72, sm, 3, 10);         // smoothed   [72, 135)
    write_posenc(row, 135, var, 3, 10);       // variance   [135, 198)

    // ---- multiresolution hash grid -> [198, 230) -------------------------
    float xn0 = (x0 * (1.0f / 1.5f) + 1.0f) * 0.5f;
    float xn1 = (x1 * (1.0f / 1.5f) + 1.0f) * 0.5f;
    float xn2 = (x2 * (1.0f / 1.5f) + 1.0f) * 0.5f;
    for (int l = 0; l < NLEVELS; ++l) {
        float scale = exp2f(4.0f + (float)l * (7.0f / 15.0f)) - 1.0f;
        int res = (int)ceilf(scale) + 1;
        bool dense = ((long long)res * res * res) <= (long long)HASHMAP;
        float p0 = xn0 * scale + 0.5f, p1 = xn1 * scale + 0.5f, p2 = xn2 * scale + 0.5f;
        float g0 = floorf(p0), g1 = floorf(p1), g2 = floorf(p2);
        float f0 = p0 - g0, f1 = p1 - g1, f2 = p2 - g2;
        unsigned u0 = (unsigned)g0, u1 = (unsigned)g1, u2 = (unsigned)g2;
        float a0 = 0.f, a1 = 0.f;
        const float* tb = table + (size_t)l * HASHMAP * 2;
        #pragma unroll
        for (int c = 0; c < 8; ++c) {
            unsigned b0 = c & 1, b1 = (c >> 1) & 1, b2 = (c >> 2) & 1;
            unsigned gg0 = u0 + b0, gg1 = u1 + b1, gg2 = u2 + b2;
            unsigned idx;
            if (dense)
                idx = gg0 + gg1 * (unsigned)res + gg2 * (unsigned)(res * res);
            else
                idx = ((gg0 * 1u) ^ (gg1 * 2654435761u) ^ (gg2 * 805459861u)) & (HASHMAP - 1u);
            float w = (b0 ? f0 : 1.0f - f0) * (b1 ? f1 : 1.0f - f1) * (b2 ? f2 : 1.0f - f2);
            const float* e = tb + (size_t)idx * 2;
            a0 += w * e[0];
            a1 += w * e[1];
        }
        row[198 + 2 * l] = (bf16_t)a0;
        row[199 + 2 * l] = (bf16_t)a1;
    }
    // zero pad K: 230..255
    for (int c = 230; c < 256; ++c) row[c] = (bf16_t)0.0f;
}

// ---------------------------------------------------------------------------
// Fused MLP
// ---------------------------------------------------------------------------
struct MlpArgs {
    const bf16_t* W[9];
    const float*  b[9];
};

// Branch-free Softplus(beta=100)/100:
//   softplus(t) = max(t,0) + log1p(exp(-|t|));  exp arg <= 0, never overflows.
__device__ __forceinline__ float softplus100(float v)
{
    float t = 100.0f * v;
    float sp = fmaxf(t, 0.0f) + __logf(1.0f + __expf(-fabsf(t)));
    return sp * 0.01f;
}

// One GEMM layer on a 64x256 tile: D = act(src @ W^T + b) * postScale
// src: LDS bf16 [64][256]; W: global bf16 [Npad][256] (row n = output feature)
// dstL: LDS bf16 [64][256]  (FINAL: dstG global f32 rows of 257)
template <int NTILES, int REALN, bool ACT, bool FINAL>
__device__ __forceinline__ void mlp_layer(
    const bf16_t* __restrict__ W, const float* __restrict__ bias,
    const bf16_t* __restrict__ src, bf16_t* __restrict__ dstL,
    float* __restrict__ dstG, float postScale)
{
    constexpr bool GUARD = (REALN < NTILES * 16);   // partial last tiles?

    const int lane  = threadIdx.x & 31;
    const int wave  = threadIdx.x >> 5;
    const int rt    = wave & 3;        // row-tile (4 x 16 rows = 64)
    const int cg    = wave >> 2;       // column group (0/1)
    const int laneN = lane & 15;
    const int hi    = (lane >> 4) & 1;
    const int row   = rt * 16 + laneN; // A-matrix row handled by this lane

    // Preload all 8 A-fragments (K = 256) for this wave's row tile.
    v16bf Afr[8];
    #pragma unroll
    for (int kt = 0; kt < 8; ++kt) {
        int off = kt * 32 + hi * 16;
        v8bf lo = *(const v8bf*)(src + row * 256 + off);
        v8bf hh = *(const v8bf*)(src + row * 256 + off + 8);
        #pragma unroll
        for (int i = 0; i < 8; ++i) { Afr[kt][i] = lo[i]; Afr[kt][8 + i] = hh[i]; }
    }

    #pragma unroll 1
    for (int nt = cg; nt < NTILES; nt += 2) {
        const int n = nt * 16 + laneN;
        v8f acc = {0.f, 0.f, 0.f, 0.f, 0.f, 0.f, 0.f, 0.f};
        #pragma unroll
        for (int kt = 0; kt < 8; ++kt) {
            // B fragment: 16 contiguous bf16 of W[n][kt*32 + hi*16 ..]
            v16bf Bfr = *(const v16bf*)(W + (size_t)n * 256 + kt * 32 + hi * 16);
            acc = __builtin_amdgcn_wmma_f32_16x16x32_bf16(
                false, Afr[kt], false, Bfr, (short)0, acc, false, false);
        }
        float bv;
        if constexpr (GUARD) bv = (n < REALN) ? bias[n] : 0.0f;
        else                 bv = bias[n];
        #pragma unroll
        for (int r = 0; r < 8; ++r) {
            int m = rt * 16 + hi * 8 + r;
            float v = acc[r] + bv;
            if constexpr (ACT) v = softplus100(v);
            v *= postScale;
            if constexpr (FINAL) {
                if (!GUARD || n < REALN) dstG[(size_t)m * 257 + n] = v;
            } else {
                if (!GUARD || n < REALN) dstL[m * 256 + n] = (bf16_t)v;
            }
        }
    }
}

__global__ __launch_bounds__(256) void mlp_kernel(
    const bf16_t* __restrict__ xin_g,  // [16384][256] bf16
    float* __restrict__ out,           // [16384][257] f32
    MlpArgs args)
{
    extern __shared__ bf16_t smem[];
    bf16_t* XIN = smem;                // 64*256
    bf16_t* XA  = smem + 64 * 256;
    bf16_t* XB  = smem + 2 * 64 * 256;
    const int tid = threadIdx.x;

    // Stage the 64x256 bf16 x_in tile into LDS (16B vectors).
    {
        const uint4* src = (const uint4*)(xin_g + (size_t)blockIdx.x * 64 * 256);
        uint4* dst = (uint4*)XIN;
        #pragma unroll
        for (int i = tid; i < (64 * 256 * 2) / 16; i += 256) dst[i] = src[i];
    }
    __syncthreads();

    mlp_layer<16, 256, true, false>(args.W[0], args.b[0], XIN, XA, nullptr, 1.0f);
    __syncthreads();
    mlp_layer<16, 256, true, false>(args.W[1], args.b[1], XA, XB, nullptr, 1.0f);
    __syncthreads();
    mlp_layer<16, 256, true, false>(args.W[2], args.b[2], XB, XA, nullptr, 1.0f);
    __syncthreads();
    // Layer 3: 256 -> 26 (padded N=32), scaled by 1/sqrt(2) into XB[:,0:26]
    mlp_layer<2, 26, true, false>(args.W[3], args.b[3], XA, XB, nullptr, INVSQRT2);
    // Skip-concat: XB[:,26:256] = XIN[:,0:230] / sqrt(2)
    for (int i = tid; i < 64 * 230; i += 256) {
        int r = i / 230, c = i - r * 230;
        XB[r * 256 + 26 + c] = (bf16_t)((float)XIN[r * 256 + c] * INVSQRT2);
    }
    __syncthreads();
    mlp_layer<16, 256, true, false>(args.W[4], args.b[4], XB, XA, nullptr, 1.0f);
    __syncthreads();
    mlp_layer<16, 256, true, false>(args.W[5], args.b[5], XA, XB, nullptr, 1.0f);
    __syncthreads();
    mlp_layer<16, 256, true, false>(args.W[6], args.b[6], XB, XA, nullptr, 1.0f);
    __syncthreads();
    mlp_layer<16, 256, true, false>(args.W[7], args.b[7], XA, XB, nullptr, 1.0f);
    __syncthreads();
    // Layer 8: 256 -> 257 (padded N=272), no activation, f32 to global
    mlp_layer<17, 257, false, true>(args.W[8], args.b[8], XB, nullptr,
                                    out + (size_t)blockIdx.x * 64 * 257, 1.0f);
}

// ---------------------------------------------------------------------------
// Launch
// ---------------------------------------------------------------------------
extern "C" void kernel_launch(void* const* d_in, const int* in_sizes, int n_in,
                              void* d_out, int out_size, void* d_ws, size_t ws_size,
                              hipStream_t stream)
{
    (void)in_sizes; (void)n_in; (void)out_size; (void)ws_size;

    const float* inp       = (const float*)d_in[0];
    const float* particles = (const float*)d_in[1];
    const float* table     = (const float*)d_in[2];

    // Padded output dims per layer (N rounded up; W0 has K padded 230->256).
    static const int Nreal[9] = {256, 256, 256, 26, 256, 256, 256, 256, 257};
    static const int Kreal[9] = {230, 256, 256, 256, 256, 256, 256, 256, 256};
    static const int Npad[9]  = {256, 256, 256, 32, 256, 256, 256, 256, 272};

    // Workspace layout (bf16 elements): 9 padded weights, then x_in.
    // Total ~9.5 MB.
    bf16_t* base = (bf16_t*)d_ws;
    size_t off = 0;
    MlpArgs args;
    bf16_t* Wp[9];
    for (int l = 0; l < 9; ++l) {
        Wp[l] = base + off;
        off += (size_t)Npad[l] * 256;
    }
    bf16_t* xin = base + off;

    // 1) convert + pad weights to bf16
    for (int l = 0; l < 9; ++l) {
        const float* W = (const float*)d_in[3 + 2 * l];
        int total = Npad[l] * 256;
        convert_w_kernel<<<(total + 255) / 256, 256, 0, stream>>>(
            W, Wp[l], Nreal[l], Kreal[l], Npad[l]);
        args.W[l] = Wp[l];
        args.b[l] = (const float*)d_in[4 + 2 * l];
    }

    // 2) features
    feature_kernel<<<NPOINTS / 256, 256, 0, stream>>>(inp, particles, table, xin);

    // 3) fused MLP: 256 blocks x 64 rows, 96 KB dynamic LDS
    size_t lds_bytes = (size_t)3 * 64 * 256 * sizeof(bf16_t);
    mlp_kernel<<<NPOINTS / 64, 256, lds_bytes, stream>>>(xin, (float*)d_out, args);
}
